// EnhancedMotionFlow_51780125721443
// MI455X (gfx1250) — compile-verified, matrix operand order checked by reading, not run on hardware
//
#include <hip/hip_runtime.h>

// ---------------------------------------------------------------------------
// EnhancedMotionFlow fused kernel for gfx1250 (MI455X, wave32, WMMA).
// One wave == one 16-row M tile. All K>=32 GEMM layers run on
// v_wmma_f32_16x16x32_bf16; K=3 layers + temporal conv run on VALU f32.
// Operands staged in LDS in native WMMA per-lane swizzle.
// ---------------------------------------------------------------------------

typedef __attribute__((ext_vector_type(16))) __bf16 v16bf;
typedef __attribute__((ext_vector_type(8)))  float  v8f;

#define B_      256
#define T_      243
#define J_      17
#define TJ_     (T_ * J_)          // 4131
#define NROWS_  (B_ * TJ_)         // 1,057,536
#define NTILES  (NROWS_ / 16)      // 66,096
#define WAVES   8
#define THREADS (WAVES * 32)
#define NBLOCKS (NTILES / WAVES)   // 8,262

#define NW_CHUNKS 44               // 44 KB of swizzled bf16 weights
#define ACT_CHUNKS 6               // 6 KB per-wave activation staging

// Storage index inside a 512-element (32 lanes x 16 bf16) WMMA operand chunk.
// lane<16 holds K 0-7 then 16-23; lane>=16 holds K 8-15 then 24-31.
__device__ __forceinline__ int swz(int kl, int nl) {
    int half = (kl >> 3) & 1;
    int e    = ((kl >> 4) << 3) | (kl & 7);
    return (nl + (half << 4)) * 16 + e;
}

struct Params {
    const float* x;
    const float* k[4];    // k1,k8,k25,k100  (ks=2,5,5,5)
    const float* w1[4];   // (3,32)
    const float* b1[4];   // (32)
    const float* w2[4];   // (32,16)
    const float* b2[4];   // (16)
    const float *ow1, *ob1, *ow2, *ob2;   // (3,64),(64),(64,64),(64)
    const float *tw1, *tb1, *tw2, *tb2;   // (64,128),(128),(128,64),(64)
    const float* sw;                      // (4)
    float* out;                           // (NROWS,64)
};

__global__ __launch_bounds__(THREADS)
void emf_kernel(Params p) {
    __shared__ __align__(32) __bf16 s_w[NW_CHUNKS * 512];
    __shared__ __align__(32) __bf16 s_a[WAVES * ACT_CHUNKS * 512];

    const int tid = threadIdx.x;

    // ---- cooperative bf16 weight conversion into WMMA-B swizzle ----------
    for (int br = 0; br < 4; ++br) {                       // chunks 0..3: branch w2 (32x16)
        const float* W = p.w2[br];
        for (int u = tid; u < 512; u += THREADS) {
            int kl = u >> 4, nl = u & 15;
            s_w[br * 512 + swz(kl, nl)] = (__bf16)W[kl * 16 + nl];
        }
    }
    for (int nt = 0; nt < 4; ++nt)                         // chunks 4..11: ow2 (64x64)
        for (int kc = 0; kc < 2; ++kc)
            for (int u = tid; u < 512; u += THREADS) {
                int kl = u >> 4, nl = u & 15;
                s_w[(4 + nt * 2 + kc) * 512 + swz(kl, nl)] =
                    (__bf16)p.ow2[(kc * 32 + kl) * 64 + nt * 16 + nl];
            }
    for (int nt = 0; nt < 8; ++nt)                         // chunks 12..27: tw1 (64x128)
        for (int kc = 0; kc < 2; ++kc)
            for (int u = tid; u < 512; u += THREADS) {
                int kl = u >> 4, nl = u & 15;
                s_w[(12 + nt * 2 + kc) * 512 + swz(kl, nl)] =
                    (__bf16)p.tw1[(kc * 32 + kl) * 128 + nt * 16 + nl];
            }
    for (int nt = 0; nt < 4; ++nt)                         // chunks 28..43: tw2 (128x64)
        for (int kc = 0; kc < 4; ++kc)
            for (int u = tid; u < 512; u += THREADS) {
                int kl = u >> 4, nl = u & 15;
                s_w[(28 + nt * 4 + kc) * 512 + swz(kl, nl)] =
                    (__bf16)p.tw2[(kc * 32 + kl) * 64 + nt * 16 + nl];
            }
    __syncthreads();

    const int lane = tid & 31;
    const int wv   = tid >> 5;
    const int m    = lane & 15;        // row within tile (A/B column lane id)
    const int half = lane >> 4;
    const int n    = m;                // output column for C/D layout

    __bf16* act = s_a + wv * (ACT_CHUNKS * 512);
    const int row0 = (blockIdx.x * WAVES + wv) * 16;
    const int r    = row0 + m;
    const int b    = r / TJ_;
    const int rem  = r - b * TJ_;
    const int t    = rem / J_;
    const int j    = rem - t * J_;

    // ---- softmax(sw) (uniform, 4 values) ---------------------------------
    float wt[4];
    {
        float s0 = p.sw[0], s1 = p.sw[1], s2 = p.sw[2], s3 = p.sw[3];
        float mx = fmaxf(fmaxf(s0, s1), fmaxf(s2, s3));
        float e0 = __expf(s0 - mx), e1 = __expf(s1 - mx);
        float e2 = __expf(s2 - mx), e3 = __expf(s3 - mx);
        float inv = 1.0f / (e0 + e1 + e2 + e3);
        wt[0] = e0 * inv; wt[1] = e1 * inv; wt[2] = e2 * inv; wt[3] = e3 * inv;
    }

    // ---- stage 0 (VALU): x taps, motion conv, K=3 layers -> LDS bf16 -----
    float xv[5][3];
#pragma unroll
    for (int k = 0; k < 5; ++k) {
        int tc = t - 2 + k;
        tc = tc < 0 ? 0 : (tc > T_ - 1 ? T_ - 1 : tc);
        const float* px = p.x + ((size_t)b * TJ_ + (size_t)tc * J_ + j) * 3;
        xv[k][0] = px[0]; xv[k][1] = px[1]; xv[k][2] = px[2];
    }
    float mv[4][3];
#pragma unroll
    for (int c = 0; c < 3; ++c)        // ks=2: taps t-1, t
        mv[0][c] = p.k[0][c] * xv[1][c] + p.k[0][3 + c] * xv[2][c];
#pragma unroll
    for (int br = 1; br < 4; ++br)     // ks=5: taps t-2..t+2
#pragma unroll
        for (int c = 0; c < 3; ++c) {
            float a = 0.0f;
#pragma unroll
            for (int k = 0; k < 5; ++k) a += p.k[br][k * 3 + c] * xv[k][c];
            mv[br][c] = a;
        }

    // branch hidden h = relu(m @ w1 + b1): lanes 0-15 do branches 0,1; 16-31 do 2,3
#pragma unroll
    for (int bb = 0; bb < 2; ++bb) {
        const float* w1 = half ? p.w1[bb + 2] : p.w1[bb];
        const float* b1 = half ? p.b1[bb + 2] : p.b1[bb];
        float q0 = half ? mv[bb + 2][0] : mv[bb][0];
        float q1 = half ? mv[bb + 2][1] : mv[bb][1];
        float q2 = half ? mv[bb + 2][2] : mv[bb][2];
        __bf16* dst = act + (half * 2 + bb) * 512;       // chunks 0..3
        for (int f = 0; f < 32; ++f) {
            float h = fmaxf(q0 * w1[f] + q1 * w1[32 + f] + q2 * w1[64 + f] + b1[f], 0.0f);
            dst[swz(f, m)] = (__bf16)h;
        }
    }
    // orig hidden hO = relu(x @ ow1 + ob1): each half computes 32 of 64 feats
    {
        __bf16* dst = act + (4 + half) * 512;            // chunks 4,5
        for (int f = 0; f < 32; ++f) {
            int fg = half * 32 + f;
            float h = fmaxf(xv[2][0] * p.ow1[fg] + xv[2][1] * p.ow1[64 + fg] +
                            xv[2][2] * p.ow1[128 + fg] + p.ob1[fg], 0.0f);
            dst[swz(f, m)] = (__bf16)h;
        }
    }
    asm volatile("" ::: "memory");

    // ---- stage 1 (WMMA): branch layer2 32->16, scale, -> combined (chunks 0,1)
    {
        v16bf A0 = *(const v16bf*)(act + 0 * 512 + lane * 16);
        v16bf A1 = *(const v16bf*)(act + 1 * 512 + lane * 16);
        v16bf A2 = *(const v16bf*)(act + 2 * 512 + lane * 16);
        v16bf A3 = *(const v16bf*)(act + 3 * 512 + lane * 16);
        asm volatile("" ::: "memory");
#pragma unroll
        for (int br = 0; br < 4; ++br) {
            v16bf A = br == 0 ? A0 : br == 1 ? A1 : br == 2 ? A2 : A3;
            v16bf Bm = *(const v16bf*)(s_w + br * 512 + lane * 16);
            v8f acc = {};
            acc = __builtin_amdgcn_wmma_f32_16x16x32_bf16(
                false, A, false, Bm, (short)0, acc, false, false);
            float bias = p.b2[br][n];
            __bf16* dst = act + (br >> 1) * 512;
            int kl = (br & 1) * 16 + n;
#pragma unroll
            for (int i = 0; i < 8; ++i) {
                float fv = fmaxf(acc[i] + bias, 0.0f) * wt[br];
                dst[swz(kl, i + half * 8)] = (__bf16)fv;
            }
        }
    }
    asm volatile("" ::: "memory");

    // ---- stage 2 (WMMA): orig layer2 64->64, keep in registers -----------
    float origr[4][8];
#pragma unroll
    for (int nt = 0; nt < 4; ++nt) {
        v8f acc = {};
#pragma unroll
        for (int kc = 0; kc < 2; ++kc) {
            v16bf A  = *(const v16bf*)(act + (4 + kc) * 512 + lane * 16);
            v16bf Bm = *(const v16bf*)(s_w + (4 + nt * 2 + kc) * 512 + lane * 16);
            acc = __builtin_amdgcn_wmma_f32_16x16x32_bf16(
                false, A, false, Bm, (short)0, acc, false, false);
        }
        float bias = p.ob2[nt * 16 + n];
#pragma unroll
        for (int i = 0; i < 8; ++i) origr[nt][i] = acc[i] + bias;
    }
    asm volatile("" ::: "memory");

    // ---- stage 3 (WMMA): trunk layer1 64->128, relu -> hT (chunks 2..5) --
#pragma unroll
    for (int nt = 0; nt < 8; ++nt) {
        v8f acc = {};
#pragma unroll
        for (int kc = 0; kc < 2; ++kc) {
            v16bf A  = *(const v16bf*)(act + kc * 512 + lane * 16);
            v16bf Bm = *(const v16bf*)(s_w + (12 + nt * 2 + kc) * 512 + lane * 16);
            acc = __builtin_amdgcn_wmma_f32_16x16x32_bf16(
                false, A, false, Bm, (short)0, acc, false, false);
        }
        float bias = p.tb1[nt * 16 + n];
        __bf16* dst = act + (2 + (nt >> 1)) * 512;
        int kl = (nt & 1) * 16 + n;
#pragma unroll
        for (int i = 0; i < 8; ++i) {
            float h = fmaxf(acc[i] + bias, 0.0f);
            dst[swz(kl, i + half * 8)] = (__bf16)h;
        }
    }
    asm volatile("" ::: "memory");

    // ---- stage 4 (WMMA): trunk layer2 128->64, relu, + orig, store -------
#pragma unroll
    for (int nt = 0; nt < 4; ++nt) {
        v8f acc = {};
#pragma unroll
        for (int kc = 0; kc < 4; ++kc) {
            v16bf A  = *(const v16bf*)(act + (2 + kc) * 512 + lane * 16);
            v16bf Bm = *(const v16bf*)(s_w + (28 + nt * 4 + kc) * 512 + lane * 16);
            acc = __builtin_amdgcn_wmma_f32_16x16x32_bf16(
                false, A, false, Bm, (short)0, acc, false, false);
        }
        float bias = p.tb2[nt * 16 + n];
#pragma unroll
        for (int i = 0; i < 8; ++i) {
            float val = origr[nt][i] + fmaxf(acc[i] + bias, 0.0f);
            p.out[(size_t)(row0 + i + half * 8) * 64 + nt * 16 + n] = val;
        }
    }
}

extern "C" void kernel_launch(void* const* d_in, const int* in_sizes, int n_in,
                              void* d_out, int out_size, void* d_ws, size_t ws_size,
                              hipStream_t stream) {
    (void)in_sizes; (void)n_in; (void)out_size; (void)d_ws; (void)ws_size;
    Params p;
    p.x = (const float*)d_in[0];
    // dict order: per scale s in {1,8,25,100}: k, w1, b1, w2, b2
    for (int s = 0; s < 4; ++s) {
        p.k[s]  = (const float*)d_in[1 + s * 5 + 0];
        p.w1[s] = (const float*)d_in[1 + s * 5 + 1];
        p.b1[s] = (const float*)d_in[1 + s * 5 + 2];
        p.w2[s] = (const float*)d_in[1 + s * 5 + 3];
        p.b2[s] = (const float*)d_in[1 + s * 5 + 4];
    }
    p.ow1 = (const float*)d_in[21];
    p.ob1 = (const float*)d_in[22];
    p.ow2 = (const float*)d_in[23];
    p.ob2 = (const float*)d_in[24];
    p.tw1 = (const float*)d_in[25];
    p.tb1 = (const float*)d_in[26];
    p.tw2 = (const float*)d_in[27];
    p.tb2 = (const float*)d_in[28];
    p.sw  = (const float*)d_in[29];
    p.out = (float*)d_out;

    emf_kernel<<<NBLOCKS, THREADS, 0, stream>>>(p);
}
